// Transformer_4750233829448
// MI455X (gfx1250) — matrix operand order repaired and checked by reading, not compiled, not run on hardware
//
#include <hip/hip_runtime.h>
#include <hip/hip_bf16.h>

typedef _Float16 half_t;
typedef __attribute__((ext_vector_type(16))) _Float16 v16h;
typedef __attribute__((ext_vector_type(8)))  _Float16 v8h;
typedef __attribute__((ext_vector_type(8)))  float    v8f;

// ---------------- problem constants ----------------
static constexpr int Bc = 2, Sc = 1024, Dc = 1024, Hc = 16, Fc = 4096, Lc = 6, Vc = 32000;
static constexpr int DKc = Dc / Hc;                  // 64
static constexpr long long BSc = (long long)Bc * Sc; // 2048

// ---------------- WMMA helpers ----------------
__device__ __forceinline__ v8f wmma32(v16h a, v16h b, v8f c) {
  return __builtin_amdgcn_wmma_f32_16x16x32_f16(false, a, false, b, (short)0, c, false, false);
}

// A fragment (16x32 f16): lanes 0-15 -> M=lane, elems 0-7=K0..7, 8-15=K16..23;
// lanes 16-31 -> elems 0-7=K8..15, 8-15=K24..31.
__device__ __forceinline__ v16h ld_frag_a(const half_t* rowbase, int hi) {
  union { v16h v; v8h h[2]; } u;
  u.h[0] = *(const v8h*)(rowbase + hi * 8);
  u.h[1] = *(const v8h*)(rowbase + 16 + hi * 8);
  return u.v;
}
// B fragment (32x16 f16) from LDS stored N-major [n][k]: lane = column,
// elems 0-7 -> K = hi*8 + i, elems 8-15 -> K = 16 + hi*8 + (i-8) (matches A).
__device__ __forceinline__ v16h ld_frag_b(const half_t* colbase, int hi) {
  union { v16h v; v8h h[2]; } u;
  u.h[0] = *(const v8h*)(colbase + hi * 8);
  u.h[1] = *(const v8h*)(colbase + 16 + hi * 8);
  return u.v;
}

// ---------------- CDNA5 async global->LDS (ASYNCcnt path) ----------------
__device__ __forceinline__ void async_ld_b128(unsigned lds_addr, const void* gaddr) {
  // VDST = LDS byte address VGPR, VADDR = 64-bit global address
  asm volatile("global_load_async_to_lds_b128 %0, %1, off"
               :: "v"(lds_addr), "v"(gaddr) : "memory");
}
__device__ __forceinline__ void wait_async0() {
#if __has_builtin(__builtin_amdgcn_s_wait_asynccnt)
  __builtin_amdgcn_s_wait_asynccnt(0);
#else
  asm volatile("s_wait_asynccnt 0x0" ::: "memory");
#endif
}
__device__ __forceinline__ unsigned lds_off(const void* p) {
  return (unsigned)(size_t)p;  // low 32 bits of generic shared pointer = LDS address
}

// =================================================================
// Kernel A: 128x128 block tile, 256 threads (8 waves, 32x64 wave tile),
// K staged 64 at a time, double-buffered LDS fed by async global->LDS.
// Requires M%128==0, N%128==0, K%64==0.
// =================================================================
__global__ __launch_bounds__(256)
void wmma_gemm128_kernel(const half_t* __restrict__ A, const half_t* __restrict__ Bm,
                         const float* __restrict__ bias,
                         float* __restrict__ Cf, half_t* __restrict__ Ch,
                         int M, int N, int K, int lda, int ldb, int ldc,
                         int transB, int relu, float alpha, int Hh,
                         long long sAb, long long sAh, long long sBb, long long sBh,
                         long long sCb, long long sCh)
{
  __shared__ __align__(16) half_t As[2][128][72];  // [m][k], row 144B (16B aligned)
  __shared__ __align__(16) half_t Bs[2][128][72];  // [n][k] (N-major)

  const int z = blockIdx.z;
  const int bb = z / Hh, hh = z % Hh;
  const half_t* Ab = A + (size_t)bb * sAb + (size_t)hh * sAh;
  const half_t* Bb = Bm + (size_t)bb * sBb + (size_t)hh * sBh;
  const size_t offC = (size_t)bb * sCb + (size_t)hh * sCh;

  const int m0 = blockIdx.y * 128, n0 = blockIdx.x * 128;
  const int tid = threadIdx.x, lane = tid & 31, wave = tid >> 5;
  const int wm = (wave & 3) * 32;        // 4 waves along M
  const int wn = (wave >> 2) * 64;       // 2 waves along N
  const int l = lane & 15, hi = lane >> 4;

  // staging coordinates: each thread owns 32 contiguous halfs of one row
  const int ar = tid >> 1, ak = (tid & 1) * 32;     // 128 rows x 2 segments of 32 halfs
  const int bkr = tid >> 3, bnn = (tid & 7) * 16;   // non-transB scatter coords

  v8f acc[2][4] = {};

  auto stage = [&](int buf, int kk) {
    { // A tile 128x64, fully async (4 x b128 per thread)
      const half_t* g = Ab + (size_t)(m0 + ar) * lda + kk + ak;
      unsigned d = lds_off(&As[buf][ar][ak]);
#pragma unroll
      for (int c = 0; c < 4; ++c) async_ld_b128(d + 16 * c, g + 8 * c);
    }
    if (transB) { // B^T rows contiguous along k -> async
      const half_t* g = Bb + (size_t)(n0 + ar) * ldb + kk + ak;
      unsigned d = lds_off(&Bs[buf][ar][ak]);
#pragma unroll
      for (int c = 0; c < 4; ++c) async_ld_b128(d + 16 * c, g + 8 * c);
    } else {      // row-major [K,N]: read along n, transpose-scatter into [n][k]
#pragma unroll
      for (int s = 0; s < 2; ++s) {
        int kr = bkr + 32 * s;
        const v8h* gp = (const v8h*)(Bb + (size_t)(kk + kr) * ldb + n0 + bnn);
        v8h q0 = gp[0], q1 = gp[1];
#pragma unroll
        for (int i = 0; i < 8; ++i) {
          Bs[buf][bnn + i][kr] = q0[i];
          Bs[buf][bnn + 8 + i][kr] = q1[i];
        }
      }
    }
  };

  const int nk = K >> 6;
  stage(0, 0);
  for (int it = 0; it < nk; ++it) {
    const int cur = it & 1;
    wait_async0();       // our async loads for buf[cur] complete
    __syncthreads();     // all waves' loads visible; all done reading buf[cur^1]
    if (it + 1 < nk) stage(cur ^ 1, (it + 1) << 6);

#pragma unroll
    for (int kh2 = 0; kh2 < 2; ++kh2) {  // two 16x16x32 steps within the staged 64
      const int ko = kh2 * 32;
      v16h a0 = ld_frag_a(&As[cur][wm + l][ko], hi);
      v16h a1 = ld_frag_a(&As[cur][wm + 16 + l][ko], hi);
      v16h b0 = ld_frag_b(&Bs[cur][wn + l][ko], hi);
      v16h b1 = ld_frag_b(&Bs[cur][wn + 16 + l][ko], hi);
      v16h b2 = ld_frag_b(&Bs[cur][wn + 32 + l][ko], hi);
      v16h b3 = ld_frag_b(&Bs[cur][wn + 48 + l][ko], hi);
      acc[0][0] = wmma32(a0, b0, acc[0][0]);
      acc[0][1] = wmma32(a0, b1, acc[0][1]);
      acc[0][2] = wmma32(a0, b2, acc[0][2]);
      acc[0][3] = wmma32(a0, b3, acc[0][3]);
      acc[1][0] = wmma32(a1, b0, acc[1][0]);
      acc[1][1] = wmma32(a1, b1, acc[1][1]);
      acc[1][2] = wmma32(a1, b2, acc[1][2]);
      acc[1][3] = wmma32(a1, b3, acc[1][3]);
    }
  }

#pragma unroll
  for (int t = 0; t < 2; ++t) {
#pragma unroll
    for (int u = 0; u < 4; ++u) {
      int col = n0 + wn + u * 16 + l;
      float bv = bias ? bias[col] : 0.f;
#pragma unroll
      for (int r = 0; r < 8; ++r) {
        int row = m0 + wm + t * 16 + hi * 8 + r;
        float v = acc[t][u][r] * alpha + bv;
        if (relu) v = v > 0.f ? v : 0.f;
        size_t idx = offC + (size_t)row * ldc + col;
        if (Cf) Cf[idx] = v;
        if (Ch) Ch[idx] = (half_t)v;
      }
    }
  }
}

// =================================================================
// Kernel B: 64x64 block tile, 128 threads (4 waves, 32x32 wave tile),
// double-buffered async pipeline, K staged 32 at a time.
// Used where N%128 != 0 (attention context GEMM, N=64). K%32==0.
// =================================================================
__global__ __launch_bounds__(128)
void wmma_gemm64_kernel(const half_t* __restrict__ A, const half_t* __restrict__ Bm,
                        const float* __restrict__ bias,
                        float* __restrict__ Cf, half_t* __restrict__ Ch,
                        int M, int N, int K, int lda, int ldb, int ldc,
                        int transB, int relu, float alpha, int Hh,
                        long long sAb, long long sAh, long long sBb, long long sBh,
                        long long sCb, long long sCh)
{
  __shared__ __align__(16) half_t As[2][64][40];
  __shared__ __align__(16) half_t Bs[2][64][40];

  const int z = blockIdx.z;
  const int bb = z / Hh, hh = z % Hh;
  const half_t* Ab = A + (size_t)bb * sAb + (size_t)hh * sAh;
  const half_t* Bb = Bm + (size_t)bb * sBb + (size_t)hh * sBh;
  const size_t offC = (size_t)bb * sCb + (size_t)hh * sCh;

  const int m0 = blockIdx.y * 64, n0 = blockIdx.x * 64;
  const int tid = threadIdx.x, lane = tid & 31, wave = tid >> 5;
  const int wm = (wave >> 1) * 32, wn = (wave & 1) * 32;
  const int l = lane & 15, hi = lane >> 4;

  const int ar = tid >> 1, ak = (tid & 1) * 16;    // 64 rows x 2 segs of 16 halfs
  const int bkr = tid >> 2, bnn = (tid & 3) * 16;  // non-transB scatter coords

  v8f acc[2][2] = {};

  auto stage = [&](int buf, int kk) {
    { const half_t* g = Ab + (size_t)(m0 + ar) * lda + kk + ak;
      unsigned d = lds_off(&As[buf][ar][ak]);
      async_ld_b128(d, g);
      async_ld_b128(d + 16, g + 8);
    }
    if (transB) {
      const half_t* g = Bb + (size_t)(n0 + ar) * ldb + kk + ak;
      unsigned d = lds_off(&Bs[buf][ar][ak]);
      async_ld_b128(d, g);
      async_ld_b128(d + 16, g + 8);
    } else {
      const v8h* gp = (const v8h*)(Bb + (size_t)(kk + bkr) * ldb + n0 + bnn);
      v8h q0 = gp[0], q1 = gp[1];
#pragma unroll
      for (int i = 0; i < 8; ++i) {
        Bs[buf][bnn + i][bkr] = q0[i];
        Bs[buf][bnn + 8 + i][bkr] = q1[i];
      }
    }
  };

  const int nk = K >> 5;
  stage(0, 0);
  for (int it = 0; it < nk; ++it) {
    const int cur = it & 1;
    wait_async0();
    __syncthreads();
    if (it + 1 < nk) stage(cur ^ 1, (it + 1) << 5);

    v16h a0 = ld_frag_a(&As[cur][wm + l][0], hi);
    v16h a1 = ld_frag_a(&As[cur][wm + 16 + l][0], hi);
    v16h b0 = ld_frag_b(&Bs[cur][wn + l][0], hi);
    v16h b1 = ld_frag_b(&Bs[cur][wn + 16 + l][0], hi);
    acc[0][0] = wmma32(a0, b0, acc[0][0]);
    acc[0][1] = wmma32(a0, b1, acc[0][1]);
    acc[1][0] = wmma32(a1, b0, acc[1][0]);
    acc[1][1] = wmma32(a1, b1, acc[1][1]);
  }

#pragma unroll
  for (int t = 0; t < 2; ++t) {
#pragma unroll
    for (int u = 0; u < 2; ++u) {
      int col = n0 + wn + u * 16 + l;
      float bv = bias ? bias[col] : 0.f;
#pragma unroll
      for (int r = 0; r < 8; ++r) {
        int row = m0 + wm + t * 16 + hi * 8 + r;
        float v = acc[t][u][r] * alpha + bv;
        if (relu) v = v > 0.f ? v : 0.f;
        size_t idx = offC + (size_t)row * ldc + col;
        if (Cf) Cf[idx] = v;
        if (Ch) Ch[idx] = (half_t)v;
      }
    }
  }
}

// ---------------- elementwise kernels ----------------
__global__ void convert_kernel(const float* __restrict__ in, half_t* __restrict__ out, long long n) {
  long long i = (long long)blockIdx.x * blockDim.x + threadIdx.x;
  long long stride = (long long)gridDim.x * blockDim.x;
  for (; i < n; i += stride) out[i] = (half_t)in[i];
}

__global__ void embed_kernel(const int* __restrict__ tok, const float* __restrict__ emb,
                             float* __restrict__ xf, half_t* __restrict__ xh, int S, int D) {
  int row = blockIdx.x;          // b*S + s
  int pos = row % S;
  int t = tok[row];
  const float* er = emb + (size_t)t * D;
  for (int d = threadIdx.x; d < D; d += blockDim.x) {
    int de = d & ~1;
    float div = __expf(-logf(10000.f) * (float)de / (float)D);
    float ang = (float)pos * div;
    float pe = (d & 1) ? __cosf(ang) : __sinf(ang);
    float v = er[d] + pe;
    xf[(size_t)row * D + d] = v;
    xh[(size_t)row * D + d] = (half_t)v;
  }
}

__global__ __launch_bounds__(256)
void add_ln_kernel(const float* __restrict__ x, const float* __restrict__ a,
                   const float* __restrict__ g, const float* __restrict__ be,
                   float* __restrict__ of, half_t* __restrict__ oh, int D) {
  __shared__ float r1[256];
  __shared__ float r2[256];
  int row = blockIdx.x, tid = threadIdx.x;
  const float* xr = x + (size_t)row * D;
  const float* ar = a ? a + (size_t)row * D : nullptr;
  float s = 0.f, ss = 0.f;
  for (int d = tid; d < D; d += 256) {
    float t = xr[d] + (ar ? ar[d] : 0.f);
    s += t; ss += t * t;
  }
  r1[tid] = s; r2[tid] = ss; __syncthreads();
  for (int w = 128; w > 0; w >>= 1) {
    if (tid < w) { r1[tid] += r1[tid + w]; r2[tid] += r2[tid + w]; }
    __syncthreads();
  }
  float mean = r1[0] / D;
  float var = r2[0] / D - mean * mean;
  float inv = rsqrtf(var + 1e-5f);
  for (int d = tid; d < D; d += 256) {
    float t = xr[d] + (ar ? ar[d] : 0.f);
    float o = (t - mean) * inv * g[d] + be[d];
    of[(size_t)row * D + d] = o;
    oh[(size_t)row * D + d] = (half_t)o;
  }
}

// masked softmax over rows of att[z][S][S] (f16 in place). S must be 1024.
__global__ __launch_bounds__(256)
void softmax_kernel(half_t* __restrict__ att, const int* __restrict__ qtok,
                    const int* __restrict__ ktok, int causal, float scale, int S, int Hh) {
  __shared__ float red[256];
  int q = blockIdx.x, z = blockIdx.y, b = z / Hh, tid = threadIdx.x;
  half_t* row = att + ((size_t)z * S + q) * (size_t)S;
  bool qvalid = qtok ? (qtok[(size_t)b * S + q] != 0) : true;
  float vals[4];
  float mx = -3e38f;
#pragma unroll
  for (int i = 0; i < 4; ++i) {
    int k = tid + i * 256;
    float s2 = (float)row[k] * scale;
    bool valid = qvalid && (!causal || k <= q) && (!ktok || ktok[(size_t)b * S + k] != 0);
    if (!valid) s2 = -1e9f;
    vals[i] = s2;
    mx = fmaxf(mx, s2);
  }
  red[tid] = mx; __syncthreads();
  for (int w = 128; w > 0; w >>= 1) { if (tid < w) red[tid] = fmaxf(red[tid], red[tid + w]); __syncthreads(); }
  float m = red[0]; __syncthreads();
  float sum = 0.f;
#pragma unroll
  for (int i = 0; i < 4; ++i) { vals[i] = __expf(vals[i] - m); sum += vals[i]; }
  red[tid] = sum; __syncthreads();
  for (int w = 128; w > 0; w >>= 1) { if (tid < w) red[tid] += red[tid + w]; __syncthreads(); }
  float inv = 1.f / red[0];
#pragma unroll
  for (int i = 0; i < 4; ++i) { int k = tid + i * 256; row[k] = (half_t)(vals[i] * inv); }
}

// ---------------- host orchestration ----------------
extern "C" void kernel_launch(void* const* d_in, const int* in_sizes, int n_in,
                              void* d_out, int out_size, void* d_ws, size_t ws_size,
                              hipStream_t stream) {
  (void)in_sizes; (void)n_in; (void)out_size; (void)ws_size;

  // input indices per setup_inputs() flattening order
  // 0 src, 1 tgt, 2 emb,
  // enc: 3 Wq 4 bq 5 Wk 6 bk 7 Wv 8 bv 9 Wo 10 bo 11 fc1 12 b1 13 fc2 14 b2 15 g1 16 be1 17 g2 18 be2
  // dec: 19 Wq 20 bq 21 Wk 22 bk 23 Wv 24 bv 25 Wo 26 bo 27 cWq 28 cbq 29 cWk 30 cbk 31 cWv 32 cbv
  //      33 cWo 34 cbo 35 fc1 36 b1 37 fc2 38 b2 39 g1 40 be1 41 g2 42 be2 43 g3 44 be3
  // 45 fc_w, 46 fc_b
  auto FP = [&](int i) { return (const float*)d_in[i]; };
  const int* src = (const int*)d_in[0];
  const int* tgt = (const int*)d_in[1];
  const float* emb = FP(2);

  const int B = Bc, S = Sc, D = Dc, H = Hc, F = Fc, Lx = Lc, V = Vc, DK = DKc;
  const long long BS = BSc;
  const long long DD = (long long)D * D, DF = (long long)D * F, DV = (long long)D * V;

  // workspace bump allocator (256B aligned)
  char* wsb = (char*)d_ws;
  size_t off = 0;
  auto alloc = [&](size_t bytes) -> void* {
    void* p = wsb + off;
    off = (off + bytes + 255) & ~(size_t)255;
    return p;
  };
  half_t* wbuf = (half_t*)alloc((size_t)DV * sizeof(half_t));        // reused weight f16 buffer
  float*  xf   = (float*)alloc((size_t)BS * D * sizeof(float));
  half_t* xh   = (half_t*)alloc((size_t)BS * D * sizeof(half_t));
  float*  yf   = (float*)alloc((size_t)BS * D * sizeof(float));
  half_t* yh   = (half_t*)alloc((size_t)BS * D * sizeof(half_t));
  half_t* qh   = (half_t*)alloc((size_t)BS * D * sizeof(half_t));
  half_t* kh   = (half_t*)alloc((size_t)BS * D * sizeof(half_t));
  half_t* vh   = (half_t*)alloc((size_t)BS * D * sizeof(half_t));
  half_t* oh   = (half_t*)alloc((size_t)BS * D * sizeof(half_t));
  float*  af   = (float*)alloc((size_t)BS * D * sizeof(float));
  half_t* att  = (half_t*)alloc((size_t)B * H * S * S * sizeof(half_t));
  half_t* ffh  = (half_t*)alloc((size_t)BS * F * sizeof(half_t));

  auto conv = [&](const float* s, half_t* d, long long n) {
    convert_kernel<<<dim3(2048), 256, 0, stream>>>(s, d, n);
  };
  auto gemm = [&](const half_t* A, const half_t* Bm, const float* bias,
                  float* Cf, half_t* Ch, int M, int N, int K,
                  int lda, int ldb, int ldc, int transB, int relu,
                  int Z, int Hh, long long sAb, long long sAh, long long sBb, long long sBh,
                  long long sCb, long long sCh) {
    if ((M % 128 == 0) && (N % 128 == 0) && (K % 64 == 0)) {
      dim3 g(N / 128, M / 128, Z);
      wmma_gemm128_kernel<<<g, 256, 0, stream>>>(A, Bm, bias, Cf, Ch, M, N, K, lda, ldb, ldc,
                                                 transB, relu, 1.0f, Hh, sAb, sAh, sBb, sBh, sCb, sCh);
    } else {
      dim3 g(N / 64, M / 64, Z);
      wmma_gemm64_kernel<<<g, 128, 0, stream>>>(A, Bm, bias, Cf, Ch, M, N, K, lda, ldb, ldc,
                                                transB, relu, 1.0f, Hh, sAb, sAh, sBb, sBh, sCb, sCh);
    }
  };
  const long long SD = (long long)S * D, SS = (long long)S * S, HSS = (long long)H * S * S;
  const float iscale = 1.0f / 8.0f;  // 1/sqrt(DK)

  auto mha = [&](half_t* xq_h, half_t* xkv_h,
                 const float* Wq, const float* bq, const float* Wk, const float* bk,
                 const float* Wv, const float* bv, const float* Wo, const float* bo,
                 const int* qtok, const int* ktok, int causal,
                 float* resid_f, half_t* resid_h, const float* g, const float* be) {
    conv(Wq, wbuf, DD);
    gemm(xq_h, wbuf, bq, nullptr, qh, BS, D, D, D, D, D, 0, 0, 1, 1, 0, 0, 0, 0, 0, 0);
    conv(Wk, wbuf, DD);
    gemm(xkv_h, wbuf, bk, nullptr, kh, BS, D, D, D, D, D, 0, 0, 1, 1, 0, 0, 0, 0, 0, 0);
    conv(Wv, wbuf, DD);
    gemm(xkv_h, wbuf, bv, nullptr, vh, BS, D, D, D, D, D, 0, 0, 1, 1, 0, 0, 0, 0, 0, 0);
    // scores[z] = Q_h (S x DK) x K_h^T (DK x S)
    gemm(qh, kh, nullptr, nullptr, att, S, S, DK, D, D, S, 1, 0,
         B * H, H, SD, DK, SD, DK, HSS, SS);
    softmax_kernel<<<dim3(S, B * H), 256, 0, stream>>>(att, qtok, ktok, causal, iscale, S, H);
    // context[z] = P (S x S) x V_h (S x DK) -> packed back to [B,S,D]
    gemm(att, vh, nullptr, nullptr, oh, S, DK, S, S, D, D, 0, 0,
         B * H, H, HSS, SS, SD, DK, SD, DK);
    conv(Wo, wbuf, DD);
    gemm(oh, wbuf, bo, af, nullptr, BS, D, D, D, D, D, 0, 0, 1, 1, 0, 0, 0, 0, 0, 0);
    add_ln_kernel<<<dim3((int)BS), 256, 0, stream>>>(resid_f, af, g, be, resid_f, resid_h, D);
  };

  auto ffn = [&](float* rf, half_t* rh, const float* fc1, const float* b1,
                 const float* fc2, const float* b2, const float* g, const float* be) {
    conv(fc1, wbuf, DF);
    gemm(rh, wbuf, b1, nullptr, ffh, BS, F, D, D, F, F, 0, 1, 1, 1, 0, 0, 0, 0, 0, 0);
    conv(fc2, wbuf, DF);
    gemm(ffh, wbuf, b2, af, nullptr, BS, D, F, F, D, D, 0, 0, 1, 1, 0, 0, 0, 0, 0, 0);
    add_ln_kernel<<<dim3((int)BS), 256, 0, stream>>>(rf, af, g, be, rf, rh, D);
  };

  // ---- embeddings + positional encoding ----
  embed_kernel<<<dim3((int)BS), 256, 0, stream>>>(src, emb, xf, xh, S, D);
  embed_kernel<<<dim3((int)BS), 256, 0, stream>>>(tgt, emb, yf, yh, S, D);

  // ---- encoder ----
  for (int i = 0; i < Lx; ++i) {
    mha(xh, xh,
        FP(3) + i * DD, FP(4) + i * D, FP(5) + i * DD, FP(6) + i * D,
        FP(7) + i * DD, FP(8) + i * D, FP(9) + i * DD, FP(10) + i * D,
        nullptr, src, 0, xf, xh, FP(15) + i * D, FP(16) + i * D);
    ffn(xf, xh, FP(11) + i * DF, FP(12) + i * F, FP(13) + i * DF, FP(14) + i * D,
        FP(17) + i * D, FP(18) + i * D);
  }

  // ---- decoder ----
  for (int i = 0; i < Lx; ++i) {
    // masked self-attention
    mha(yh, yh,
        FP(19) + i * DD, FP(20) + i * D, FP(21) + i * DD, FP(22) + i * D,
        FP(23) + i * DD, FP(24) + i * D, FP(25) + i * DD, FP(26) + i * D,
        tgt, nullptr, 1, yf, yh, FP(39) + i * D, FP(40) + i * D);
    // cross-attention over encoder output (xh)
    mha(yh, xh,
        FP(27) + i * DD, FP(28) + i * D, FP(29) + i * DD, FP(30) + i * D,
        FP(31) + i * DD, FP(32) + i * D, FP(33) + i * DD, FP(34) + i * D,
        nullptr, src, 0, yf, yh, FP(41) + i * D, FP(42) + i * D);
    ffn(yf, yh, FP(35) + i * DF, FP(36) + i * F, FP(37) + i * DF, FP(38) + i * D,
        FP(43) + i * D, FP(44) + i * D);
  }

  // ---- final projection to vocab (f32 output) ----
  conv(FP(45), wbuf, DV);
  gemm(yh, wbuf, FP(46), (float*)d_out, nullptr, BS, V, D, D, V, V, 0, 0,
       1, 1, 0, 0, 0, 0, 0, 0);
}
